// GotenNet_28123445854675
// MI455X (gfx1250) — matrix operand order repaired and checked by reading, not compiled, not run on hardware
//
#include <hip/hip_runtime.h>
#include <hip/hip_bf16.h>

typedef __hip_bfloat16 bf16;
typedef __attribute__((ext_vector_type(16))) __bf16 v16bf;
typedef __attribute__((ext_vector_type(8)))  __bf16 v8bf;
typedef __attribute__((ext_vector_type(8)))  float  v8f;
typedef __attribute__((ext_vector_type(4)))  unsigned int u32x4;
typedef __attribute__((ext_vector_type(8)))  int i32x8;
typedef __attribute__((ext_vector_type(4)))  int i32x4;

#define DV 128          // NODE_DIM
#define XCOMP 8         // 3 + 5 vector components
#define CUTOFF_R 5.0f

__device__ __forceinline__ float silu_f(float x) {
    return x / (1.0f + __expf(-x));
}

__device__ __forceinline__ void atomicMaxF(float* addr, float val) {
    unsigned int* ua = (unsigned int*)addr;
    unsigned int old = *ua;
    while (true) {
        float f = __uint_as_float(old);
        if (f >= val) break;
        unsigned int assumed = old;
        old = atomicCAS(ua, assumed, __float_as_uint(val));
        if (old == assumed) break;
    }
}

// ---------------------------------------------------------------------------
// Generic bf16 WMMA GEMM:  C[M][N] = act( A_f32[M][K] @ Wt^T + bias )
// Wt is stored transposed: Wt[n][k] (bf16).  Optional grouped row mapping:
// physical_row(m) = (m/grp_sz)*grp_stride + grp_off + m%grp_sz  (A and C).
//
// The 64xK bf16 weight panel for this block is DMA'd into LDS ONCE by the
// Tensor Data Mover (TENSOR_LOAD_TO_LDS, tracked by TENSORcnt), then the
// k-loop only stages A tiles (f32 -> bf16 conversion in the staging path).
// Block tile 128x64, 8 waves (wave32), each wave computes a 16x64 slice
// with four v_wmma_f32_16x16x32_bf16 per K-step.
// ---------------------------------------------------------------------------
__global__ void __launch_bounds__(256)
k_gemm(const float* __restrict__ A, const bf16* __restrict__ Wt,
       const float* __restrict__ bias, float* __restrict__ C,
       int M, int K, int N, int act, int grp_sz, int grp_stride, int grp_off)
{
    __shared__ bf16 As[128][40];      // 32 + 8 pad (bf16 elems)
    extern __shared__ bf16 BsD[];     // dynamic: 64 rows x K bf16 weight panel

    const int tid  = threadIdx.x;
    const int lane = tid & 31;
    const int wave = tid >> 5;
    const int m0   = blockIdx.x * 128;
    const int n0   = blockIdx.y * 64;

    // ---- resident weight panel: TDM async copy global -> LDS ----
#if __has_builtin(__builtin_amdgcn_tensor_load_to_lds)
    if (tid < 32) {   // wave 0 issues the tensor DMA (EXEC ignored by TDM)
        unsigned lds_base = (unsigned)(unsigned long long)(void*)&BsD[0];
        unsigned long long ga =
            (unsigned long long)(const void*)(Wt + (long)n0 * K);
        // D# group 0: count=1 | lds_addr | global_addr[56:0] | type=2
        u32x4 g0;
        g0[0] = 1u;
        g0[1] = lds_base;
        g0[2] = (unsigned)ga;
        g0[3] = ((unsigned)(ga >> 32) & 0x01FFFFFFu) | (2u << 30);
        // D# group 1: data_size=2B; tensor K x 64; tile K x 64; stride0=K
        i32x8 g1;
        g1[0] = 0x10000;                          // data_size code 1 (2 bytes)
        g1[1] = (int)(((unsigned)K & 0xFFFFu) << 16);   // tensor_dim0[15:0]
        g1[2] = (int)(64u << 16);                 // tensor_dim0[31:16]=0 | tensor_dim1=64
        g1[3] = (int)(((unsigned)K & 0xFFFFu) << 16);   // tensor_dim1 hi=0 | tile_dim0=K
        g1[4] = 64;                               // tile_dim1=64 | tile_dim2=0
        g1[5] = K;                                // tensor_dim0_stride[31:0]
        g1[6] = 0;                                // stride0 hi | stride1 lo
        g1[7] = 0;
        i32x4 gz = {0, 0, 0, 0};                  // groups 2/3 unused (2-D tile)
#if __clang_major__ >= 23
        i32x8 gz8 = {0, 0, 0, 0, 0, 0, 0, 0};
        __builtin_amdgcn_tensor_load_to_lds(g0, g1, gz, gz, gz8, 0);
#else
        __builtin_amdgcn_tensor_load_to_lds(g0, g1, gz, gz, 0);
#endif
        __builtin_amdgcn_s_wait_tensorcnt(0);
    }
    __syncthreads();
#else
    // fallback: cooperative vector copy of the weight panel
    for (int idx = tid; idx < (64 * K) / 8; idx += 256) {
        int n = (idx * 8) / K, k = (idx * 8) % K;
        *(v8bf*)&BsD[n * K + k] = *(const v8bf*)(Wt + (long)(n0 + n) * K + k);
    }
    __syncthreads();
#endif

    v8f acc[4];
#pragma unroll
    for (int j = 0; j < 4; ++j)
#pragma unroll
        for (int i = 0; i < 8; ++i) acc[j][i] = 0.0f;

    for (int k0 = 0; k0 < K; k0 += 32) {
        // ---- stage A tile (convert f32 -> bf16) ----
        {
            int r  = tid >> 1;
            int c  = (tid & 1) * 16;
            int lm = m0 + r;
            bool valid = lm < M;
            long arow = 0;
            if (valid) {
                arow = (grp_sz > 1)
                     ? (long)(lm / grp_sz) * grp_stride + grp_off + (lm % grp_sz)
                     : (long)lm;
            }
            const float* ap = A + arow * (long)K + k0 + c;
            if (valid && (k0 + 32 < K)) {
                __builtin_prefetch((const void*)(ap + 32), 0, 3);  // global_prefetch
            }
#pragma unroll
            for (int q = 0; q < 4; ++q) {
                float4 f = valid ? *(const float4*)(ap + q * 4)
                                 : make_float4(0.f, 0.f, 0.f, 0.f);
                As[r][c + q * 4 + 0] = __float2bfloat16(f.x);
                As[r][c + q * 4 + 1] = __float2bfloat16(f.y);
                As[r][c + q * 4 + 2] = __float2bfloat16(f.z);
                As[r][c + q * 4 + 3] = __float2bfloat16(f.w);
            }
        }
        __syncthreads();

        // ---- WMMA compute ----
        const int mloc = lane & 15;
        const int half = lane >> 4;
        union { v16bf v; v8bf h[2]; } afrag;
        afrag.h[0] = *(const v8bf*)&As[wave * 16 + mloc][half * 8];
        afrag.h[1] = *(const v8bf*)&As[wave * 16 + mloc][16 + half * 8];
#pragma unroll
        for (int j = 0; j < 4; ++j) {
            union { v16bf v; v8bf h[2]; } bfrag;
            const bf16* bp = &BsD[(long)(j * 16 + mloc) * K + k0 + half * 8];
            bfrag.h[0] = *(const v8bf*)bp;
            bfrag.h[1] = *(const v8bf*)(bp + 16);
            acc[j] = __builtin_amdgcn_wmma_f32_16x16x32_bf16(
                false, afrag.v, false, bfrag.v, (short)0, acc[j], false, false);
        }
        __syncthreads();
    }

    // ---- epilogue: bias + activation + store ----
    const int mloc  = lane & 15;
    const int rhalf = lane >> 4;
#pragma unroll
    for (int j = 0; j < 4; ++j) {
        int gn = n0 + j * 16 + mloc;
#pragma unroll
        for (int i = 0; i < 8; ++i) {
            int gm = m0 + wave * 16 + rhalf * 8 + i;
            if (gm < M) {
                float v = acc[j][i];
                if (bias) v += bias[gn];
                if (act == 1) v = silu_f(v);
                long crow = (grp_sz > 1)
                          ? (long)(gm / grp_sz) * grp_stride + grp_off + (gm % grp_sz)
                          : (long)gm;
                C[crow * (long)N + gn] = v;
            }
        }
    }
}

// ---------------------------------------------------------------------------
// small utility kernels
// ---------------------------------------------------------------------------
__global__ void k_prep_wt(const float* __restrict__ W, bf16* __restrict__ Wt,
                          int K, int N) {
    long i = (long)blockIdx.x * blockDim.x + threadIdx.x;
    if (i < (long)K * N) {
        int k = (int)(i / N), n = (int)(i % N);
        Wt[(long)n * K + k] = __float2bfloat16(W[i]);
    }
}

__global__ void k_fill(float* p, float v, long long n) {
    long long i = (long long)blockIdx.x * blockDim.x + threadIdx.x;
    if (i < n) p[i] = v;
}

__global__ void k_copy(float* dst, const float* src, long long n) {
    long long i = (long long)blockIdx.x * blockDim.x + threadIdx.x;
    if (i < n) dst[i] = src[i];
}

__global__ void k_edge_geom(const float* __restrict__ xe, float* r0, float* cc,
                            float* l1, float* l2, int E) {
    int e = blockIdx.x * blockDim.x + threadIdx.x;
    if (e >= E) return;
    float x = xe[e * 3 + 0], y = xe[e * 3 + 1], z = xe[e * 3 + 2];
    float r = sqrtf(x * x + y * y + z * z);
    r0[e] = r;
    float c = (r < CUTOFF_R)
            ? 0.5f * (cosf(3.14159265358979323f * r / CUTOFF_R) + 1.0f) : 0.0f;
    cc[e] = c;
    float inv = 1.0f / r;
    float ux = x * inv, uy = y * inv, uz = z * inv;
    l1[e * 3 + 0] = ux; l1[e * 3 + 1] = uy; l1[e * 3 + 2] = uz;
    const float s3 = 1.7320508075688772f;
    l2[e * 5 + 0] = s3 * ux * uy;
    l2[e * 5 + 1] = s3 * uy * uz;
    l2[e * 5 + 2] = 0.5f * (3.0f * uz * uz - 1.0f);
    l2[e * 5 + 3] = s3 * ux * uz;
    l2[e * 5 + 4] = 0.5f * s3 * (ux * ux - uy * uy);
}

__global__ void k_rbf(const float* __restrict__ r0, const float* __restrict__ cc,
                      const float* __restrict__ centers,
                      const float* __restrict__ gamma,
                      float* __restrict__ rbf0, int E) {
    long long i = (long long)blockIdx.x * blockDim.x + threadIdx.x;
    if (i >= (long long)E * 64) return;
    int e = (int)(i >> 6), c = (int)(i & 63);
    float d = __expf(-r0[e]) - centers[c];       // alpha = 5/CUTOFF = 1
    rbf0[i] = __expf(-gamma[0] * d * d) * cc[e];
}

__global__ void k_msg_scatter(const float* __restrict__ ndp,
                              const float* __restrict__ cc,
                              const float* __restrict__ a_nbr,
                              const int* __restrict__ xn,
                              const int* __restrict__ nj, const int* __restrict__ ni,
                              float* __restrict__ m_i, int E) {
    long long i = (long long)blockIdx.x * blockDim.x + threadIdx.x;
    if (i >= (long long)E * DV) return;
    int e = (int)(i >> 7), d = (int)(i & 127);
    float val = a_nbr[(long)xn[nj[e]] * DV + d] * ndp[i] * cc[e];
    atomicAdd(&m_i[(long)ni[e] * DV + d], val);
}

__global__ void k_build_hcat(const float* __restrict__ a_na,
                             const int* __restrict__ xn,
                             const float* __restrict__ m_i,
                             float* __restrict__ hcat, int N) {
    long long i = (long long)blockIdx.x * blockDim.x + threadIdx.x;
    if (i >= (long long)N * DV) return;
    int n = (int)(i >> 7), d = (int)(i & 127);
    hcat[(long)n * 256 + d]       = a_na[(long)xn[n] * DV + d];
    hcat[(long)n * 256 + 128 + d] = m_i[i];
}

// block per row, blockDim == dim (power of two), shared reduction
__global__ void k_layernorm(const float* __restrict__ x, const float* __restrict__ g,
                            const float* __restrict__ be, float* __restrict__ y,
                            int dim, int act) {
    extern __shared__ float sm[];
    long row = blockIdx.x;
    int t = threadIdx.x;
    float v = x[row * (long)dim + t];
    sm[t] = v; __syncthreads();
    for (int s = dim >> 1; s > 0; s >>= 1) {
        if (t < s) sm[t] += sm[t + s];
        __syncthreads();
    }
    float mu = sm[0] / dim; __syncthreads();
    float d = v - mu;
    sm[t] = d * d; __syncthreads();
    for (int s = dim >> 1; s > 0; s >>= 1) {
        if (t < s) sm[t] += sm[t + s];
        __syncthreads();
    }
    float var = sm[0] / dim;
    float o = d * rsqrtf(var + 1e-5f) * g[t] + be[t];
    if (act) o = silu_f(o);
    y[row * (long)dim + t] = o;
}

__global__ void k_tij_init(const float* __restrict__ h, const float* __restrict__ erp,
                           const int* __restrict__ nj, const int* __restrict__ ni,
                           float* __restrict__ t, int E) {
    long long i = (long long)blockIdx.x * blockDim.x + threadIdx.x;
    if (i >= (long long)E * DV) return;
    int e = (int)(i >> 7), d = (int)(i & 127);
    t[i] = (h[(long)ni[e] * DV + d] + h[(long)nj[e] * DV + d]) * erp[i];
}

__global__ void k_sea_logits(const float* __restrict__ qn, const float* __restrict__ kn,
                             const float* __restrict__ gE,
                             const int* __restrict__ nj, const int* __restrict__ ni,
                             float* __restrict__ a, int E) {
    long long i = (long long)blockIdx.x * blockDim.x + threadIdx.x;
    if (i >= (long long)E * 8) return;
    int e = (int)(i >> 3), hd = (int)(i & 7);
    const float* qp = qn + (long)ni[e] * DV + hd * 16;
    const float* kp = kn + (long)nj[e] * DV + hd * 16;
    const float* gp = gE + (long)e * DV + hd * 16;
    float s = 0.f;
#pragma unroll
    for (int c = 0; c < 16; ++c) s += qp[c] * kp[c] * gp[c];
    a[i] = s;
}

__global__ void k_seg_max(const float* __restrict__ a, const int* __restrict__ ni,
                          float* __restrict__ amax, int E) {
    long long i = (long long)blockIdx.x * blockDim.x + threadIdx.x;
    if (i >= (long long)E * 8) return;
    int e = (int)(i >> 3), hd = (int)(i & 7);
    atomicMaxF(&amax[(long)ni[e] * 8 + hd], a[i]);
}

__global__ void k_seg_exp_sum(float* __restrict__ a, const float* __restrict__ amax,
                              const int* __restrict__ ni, float* __restrict__ asum,
                              int E) {
    long long i = (long long)blockIdx.x * blockDim.x + threadIdx.x;
    if (i >= (long long)E * 8) return;
    int e = (int)(i >> 3), hd = (int)(i & 7);
    float ex = __expf(a[i] - amax[(long)ni[e] * 8 + hd]);
    a[i] = ex;
    atomicAdd(&asum[(long)ni[e] * 8 + hd], ex);
}

__global__ void k_seg_norm_av(const float* __restrict__ a, const float* __restrict__ asum,
                              const int* __restrict__ ni, const int* __restrict__ nj,
                              const float* __restrict__ vn, float* __restrict__ av,
                              int E, int SD) {
    long long i = (long long)blockIdx.x * blockDim.x + threadIdx.x;
    if (i >= (long long)E * SD) return;
    int e = (int)(i / SD), u = (int)(i % SD);
    int dv = SD / 8;
    int hd = u / dv;
    float w = a[(long)e * 8 + hd] / (asum[(long)ni[e] * 8 + hd] + 1e-16f)
            * rsqrtf((float)dv);
    av[i] = w * vn[(long)nj[e] * SD + u];
}

__global__ void k_scatter_h(const float* __restrict__ sea, const float* __restrict__ rs,
                            const float* __restrict__ sfac, const float* __restrict__ cc,
                            const int* __restrict__ nj, const int* __restrict__ ni,
                            float* __restrict__ h, int E, int SD) {
    long long i = (long long)blockIdx.x * blockDim.x + threadIdx.x;
    if (i >= (long long)E * DV) return;
    int e = (int)(i >> 7), d = (int)(i & 127);
    long b = (long)e * SD + d;
    float o = sea[b] + rs[b] * sfac[(long)nj[e] * SD + d] * cc[e];
    atomicAdd(&h[(long)ni[e] * DV + d], o);
}

__global__ void k_scatter_X(const float* __restrict__ sea, const float* __restrict__ rs,
                            const float* __restrict__ sfac, const float* __restrict__ cc,
                            const float* __restrict__ l1e, const float* __restrict__ l2e,
                            const float* __restrict__ Xold,
                            const int* __restrict__ nj, const int* __restrict__ ni,
                            float* __restrict__ Xnew, int E, int SD, int first) {
    long long i = (long long)blockIdx.x * blockDim.x + threadIdx.x;
    if (i >= (long long)E * (XCOMP * DV)) return;
    int e    = (int)(i >> 10);
    int rest = (int)(i & 1023);
    int p = rest >> 7, d = rest & 127;
    int deg = (p < 3) ? 0 : 1;
    float rl = (deg == 0) ? l1e[(long)e * 3 + p] : l2e[(long)e * 5 + (p - 3)];
    float c = cc[e];
    int jn = nj[e];
    long b1 = (long)e * SD + (1 + deg) * DV + d;
    float o1 = sea[b1] + rs[b1] * sfac[(long)jn * SD + (1 + deg) * DV + d] * c;
    float m = o1 * rl;
    if (!first) {
        long b2 = (long)e * SD + (3 + deg) * DV + d;
        float o2 = sea[b2] + rs[b2] * sfac[(long)jn * SD + (3 + deg) * DV + d] * c;
        m += o2 * Xold[(long)jn * (XCOMP * DV) + rest];
    }
    atomicAdd(&Xnew[(long)ni[e] * (XCOMP * DV) + rest], m);
}

__global__ void k_wij(const float* __restrict__ eqn, const float* __restrict__ ekn,
                      const float* __restrict__ l1e, const float* __restrict__ l2e,
                      const int* __restrict__ nj, const int* __restrict__ ni,
                      float* __restrict__ wij, int E) {
    long long i = (long long)blockIdx.x * blockDim.x + threadIdx.x;
    if (i >= (long long)E * DV) return;
    int e = (int)(i >> 7), d = (int)(i & 127);
    long bi = (long)ni[e] * (XCOMP * DV) + d;
    long bj = (long)nj[e] * (XCOMP * DV) + d;
    float eq[8], ek[8];
#pragma unroll
    for (int p = 0; p < 8; ++p) {
        eq[p] = eqn[bi + (long)p * DV];
        ek[p] = ekn[bj + (long)p * DV];
    }
    float acc = 0.f;
    {   // degree 1 (3 comps) -- rejection is sign invariant: rej(x,-r)==rej(x,r)
        float r[3] = { l1e[(long)e * 3 + 0], l1e[(long)e * 3 + 1], l1e[(long)e * 3 + 2] };
        float dq = 0.f, dk = 0.f;
#pragma unroll
        for (int p = 0; p < 3; ++p) { dq += eq[p] * r[p]; dk += ek[p] * r[p]; }
#pragma unroll
        for (int p = 0; p < 3; ++p)
            acc += (eq[p] - dq * r[p]) * (ek[p] - dk * r[p]);
    }
    {   // degree 2 (5 comps)
        float r[5];
#pragma unroll
        for (int p = 0; p < 5; ++p) r[p] = l2e[(long)e * 5 + p];
        float dq = 0.f, dk = 0.f;
#pragma unroll
        for (int p = 0; p < 5; ++p) { dq += eq[3 + p] * r[p]; dk += ek[3 + p] * r[p]; }
#pragma unroll
        for (int p = 0; p < 5; ++p)
            acc += (eq[3 + p] - dq * r[p]) * (ek[3 + p] - dk * r[p]);
    }
    wij[i] = acc;
}

__global__ void k_htr_apply(float* __restrict__ t, const float* __restrict__ ww,
                            const float* __restrict__ th2, long long n) {
    long long i = (long long)blockIdx.x * blockDim.x + threadIdx.x;
    if (i < n) t[i] += ww[i] * silu_f(th2[i]);
}

__global__ void k_l2n(const float* __restrict__ Xvu, float* __restrict__ l2n, int N) {
    long long i = (long long)blockIdx.x * blockDim.x + threadIdx.x;
    if (i >= (long long)N * DV) return;
    int n = (int)(i >> 7), d = (int)(i & 127);
    float s = 1e-8f;   // EPS
#pragma unroll
    for (int p = 0; p < 8; ++p) {
        float v = Xvu[(long)n * (XCOMP * DV) + (long)p * DV + d];
        s += v * v;
    }
    l2n[i] = sqrtf(s);
}

__global__ void k_cat(const float* __restrict__ h, const float* __restrict__ l2n,
                      float* __restrict__ cat, int N) {
    long long i = (long long)blockIdx.x * blockDim.x + threadIdx.x;
    if (i >= (long long)N * DV) return;
    int n = (int)(i >> 7), d = (int)(i & 127);
    cat[(long)n * 256 + d]       = h[i];
    cat[(long)n * 256 + 128 + d] = l2n[i];
}

__global__ void k_add_h(float* __restrict__ h, const float* __restrict__ mm, long long n) {
    long long i = (long long)blockIdx.x * blockDim.x + threadIdx.x;
    if (i < n) h[i] += mm[(i >> 7) * 256 + (i & 127)];
}

__global__ void k_xupd(float* __restrict__ X, const float* __restrict__ mm,
                       const float* __restrict__ Xvu, long long n) {
    long long i = (long long)blockIdx.x * blockDim.x + threadIdx.x;
    if (i < n) {
        long long nn = i >> 10;
        int d = (int)(i & 127);
        X[i] += mm[nn * 256 + 128 + d] * Xvu[i];
    }
}

__global__ void k_decode(const float* __restrict__ dh, const float* __restrict__ w2,
                         const float* __restrict__ b2, const int* __restrict__ batch,
                         float* __restrict__ out, int N) {
    int n = blockIdx.x * blockDim.x + threadIdx.x;
    if (n >= N) return;
    float s = b2[0];
#pragma unroll 4
    for (int d = 0; d < DV; ++d) s += dh[(long)n * DV + d] * w2[d];
    atomicAdd(&out[batch[n]], s);
}

// ---------------------------------------------------------------------------
static inline int gr(long long n) { return (int)((n + 255) / 256); }

extern "C" void kernel_launch(void* const* d_in, const int* in_sizes, int n_in,
                              void* d_out, int out_size, void* d_ws, size_t ws_size,
                              hipStream_t stream) {
    (void)n_in; (void)ws_size;
    const int N = in_sizes[2];          // 10000 nodes
    const int E = in_sizes[4] / 2;      // 160000 edges

    auto F = [&](int i) -> const float* { return (const float*)d_in[i]; };
    const int* xn    = (const int*)d_in[2];
    const int* eidx  = (const int*)d_in[4];
    const int* nj    = eidx;            // edge_index[0]
    const int* ni    = eidx + E;        // edge_index[1]
    const int* batch = (const int*)d_in[5];
    float* out = (float*)d_out;

    // params flattened after the 6 data inputs (jax tree: sorted dict keys)
    // dec: 6 l1.b, 7 l1.w, 8 l2.b, 9 l2.w
    // emb: 10 a_na, 11 a_nbr, 12 centers, 13 gamma, 14 nru_l1.b, 15 nru_l1.w,
    //      16 nru_l2.b, 17 nru_l2.w, 18 nru_ln.be, 19 nru_ln.g,
    //      20 w_erp.b, 21 w_erp.w, 22 w_ndp.b, 23 w_ndp.w
    struct EI { int m1b, m1w, m2b, m2w, lnbe, lng, vuw; };
    struct GI { int lnbe, lng, s1b, s1w, s2b, s2w, cb, cw,
                    v1b, v1w, v2b, v2w, kb, kw, qb, qw, reb, rew, rsb, rsw; };
    const EI EL[2] = { {24,25,26,27,28,29,30}, {51,52,53,54,55,56,57} };
    const GI GL[2] = { {31,32,33,34,35,36,37,38,39,40,41,42,43,44,45,46,47,48,49,50},
                       {69,70,71,72,73,74,75,76,77,78,79,80,81,82,83,84,85,86,87,88} };
    // htr (layer 1): 58 mt1.b 59 mt1.w 60 mt2.b 61 mt2.w 62 mwl.b 63 mwl.w
    //                64 mwln.be 65 mwln.g 66 vk0.w 67 vk1.w 68 vq.w
    const int SDv[2] = {384, 640};

    // ---- workspace bump allocator ----
    char* base = (char*)d_ws;
    size_t off = 0;
    auto alloc = [&](size_t bytes) -> void* {
        size_t o = (off + 255) & ~(size_t)255;
        off = o + bytes;
        return (void*)(base + o);
    };
    auto allocf = [&](long long n) -> float* { return (float*)alloc((size_t)n * 4); };

    auto prep = [&](int widx, int K, int Nn) -> bf16* {
        bf16* wt = (bf16*)alloc((size_t)K * Nn * 2);
        k_prep_wt<<<gr((long long)K * Nn), 256, 0, stream>>>(F(widx), wt, K, Nn);
        return wt;
    };
    auto gemm = [&](const float* A, const bf16* Wt, const float* bias, float* C,
                    int M, int K, int Nn, int act, int gsz, int gst, int gof) {
        dim3 g((M + 127) / 128, Nn / 64);
        size_t shmem = (size_t)64 * K * sizeof(bf16);   // resident weight panel
        k_gemm<<<g, 256, shmem, stream>>>(A, Wt, bias, C, M, K, Nn, act, gsz, gst, gof);
    };
    auto lnorm = [&](const float* x, int gidx, int beidx, float* y,
                     int rows, int dim, int act) {
        k_layernorm<<<rows, dim, dim * sizeof(float), stream>>>(
            x, F(gidx), F(beidx), y, dim, act);
    };

    // ---- weight prep (f32 -> transposed bf16) ----
    bf16* w_ndp = prep(23, 64, 128);
    bf16* w_erp = prep(21, 64, 128);
    bf16* w_n1  = prep(15, 256, 128);
    bf16* w_n2  = prep(17, 128, 128);
    bf16* w_d1  = prep(7, 128, 128);
    bf16 *wq[2], *wk[2], *wv1[2], *wv2[2], *wre[2], *wcmb[2], *wrs[2],
         *ws1[2], *ws2[2], *wm1[2], *wm2[2], *wvu[2];
    for (int l = 0; l < 2; ++l) {
        int SD = SDv[l];
        wq[l]   = prep(GL[l].qw, 128, 128);
        wk[l]   = prep(GL[l].kw, 128, 128);
        wv1[l]  = prep(GL[l].v1w, 128, 128);
        wv2[l]  = prep(GL[l].v2w, 128, SD);
        wre[l]  = prep(GL[l].rew, 128, 128);
        wcmb[l] = prep(GL[l].cw, SD, SD);
        wrs[l]  = prep(GL[l].rsw, 128, SD);
        ws1[l]  = prep(GL[l].s1w, 128, 128);
        ws2[l]  = prep(GL[l].s2w, 128, SD);
        wm1[l]  = prep(EL[l].m1w, 256, 128);
        wm2[l]  = prep(EL[l].m2w, 128, 256);
        wvu[l]  = prep(EL[l].vuw, 128, 128);
    }
    bf16* wmt1 = prep(59, 128, 128);
    bf16* wmt2 = prep(61, 128, 128);
    bf16* wmwl = prep(63, 128, 128);
    bf16* wvk0 = prep(66, 128, 128);
    bf16* wvk1 = prep(67, 128, 128);
    bf16* wvq  = prep(68, 128, 128);

    // ---- embedding ----
    float* r0   = allocf(E);
    float* ccv  = allocf(E);
    float* l1e  = allocf((long long)E * 3);
    float* l2e  = allocf((long long)E * 5);
    float* rbf0 = allocf((long long)E * 64);
    k_edge_geom<<<gr(E), 256, 0, stream>>>(F(3), r0, ccv, l1e, l2e, E);
    k_rbf<<<gr((long long)E * 64), 256, 0, stream>>>(r0, ccv, F(12), F(13), rbf0, E);

    float* ndp = allocf((long long)E * DV);
    gemm(rbf0, w_ndp, F(22), ndp, E, 64, 128, 0, 1, 1, 0);
    float* m_i = allocf((long long)N * DV);
    k_fill<<<gr((long long)N * DV), 256, 0, stream>>>(m_i, 0.f, (long long)N * DV);
    k_msg_scatter<<<gr((long long)E * DV), 256, 0, stream>>>(ndp, ccv, F(11), xn, nj, ni, m_i, E);

    float* hcat = allocf((long long)N * 256);
    k_build_hcat<<<gr((long long)N * DV), 256, 0, stream>>>(F(10), xn, m_i, hcat, N);
    float* t1 = allocf((long long)N * DV);
    gemm(hcat, w_n1, F(14), t1, N, 256, 128, 0, 1, 1, 0);
    lnorm(t1, 19, 18, t1, N, 128, 1);                 // act(ln(...))
    float* h = allocf((long long)N * DV);
    gemm(t1, w_n2, F(16), h, N, 128, 128, 0, 1, 1, 0);

    float* erp = allocf((long long)E * DV);
    gemm(rbf0, w_erp, F(20), erp, E, 64, 128, 0, 1, 1, 0);
    float* tij = allocf((long long)E * DV);
    k_tij_init<<<gr((long long)E * DV), 256, 0, stream>>>(h, erp, nj, ni, tij, E);

    // ---- GATA + EQFF layers ----
    float* X = nullptr;
    for (int l = 0; l < 2; ++l) {
        const GI& G = GL[l];
        const EI& Q = EL[l];
        const int SD = SDv[l];

        float* hln = allocf((long long)N * DV);
        lnorm(h, G.lng, G.lnbe, hln, N, 128, 0);

        if (l == 1) {   // htr: t_ij += mlp_w_l(ln(w_ij)) * silu(mlp_t(t_ij))
            float* eqn = allocf((long long)N * XCOMP * DV);
            float* ekn = allocf((long long)N * XCOMP * DV);
            gemm(X, wvq, nullptr, eqn, N * 8, 128, 128, 0, 1, 1, 0);
            gemm(X, wvk0, nullptr, ekn, N * 3, 128, 128, 0, 3, 8, 0);
            gemm(X, wvk1, nullptr, ekn, N * 5, 128, 128, 0, 5, 8, 3);
            float* wij = allocf((long long)E * DV);
            k_wij<<<gr((long long)E * DV), 256, 0, stream>>>(eqn, ekn, l1e, l2e, nj, ni, wij, E);
            float* wln = allocf((long long)E * DV);
            lnorm(wij, 65, 64, wln, E, 128, 0);
            float* ww = allocf((long long)E * DV);
            gemm(wln, wmwl, F(62), ww, E, 128, 128, 0, 1, 1, 0);
            float* th  = allocf((long long)E * DV);
            float* th2 = allocf((long long)E * DV);
            gemm(tij, wmt1, F(58), th, E, 128, 128, 1, 1, 1, 0);
            gemm(th, wmt2, F(60), th2, E, 128, 128, 0, 1, 1, 0);
            k_htr_apply<<<gr((long long)E * DV), 256, 0, stream>>>(tij, ww, th2, (long long)E * DV);
        }

        // sea attention
        float* qn = allocf((long long)N * DV);
        float* kn = allocf((long long)N * DV);
        float* vh = allocf((long long)N * DV);
        float* vn = allocf((long long)N * SD);
        gemm(hln, wq[l], F(G.qb), qn, N, 128, 128, 0, 1, 1, 0);
        gemm(hln, wk[l], F(G.kb), kn, N, 128, 128, 0, 1, 1, 0);
        gemm(hln, wv1[l], F(G.v1b), vh, N, 128, 128, 1, 1, 1, 0);
        gemm(vh, wv2[l], F(G.v2b), vn, N, 128, SD, 0, 1, 1, 0);
        float* gE = allocf((long long)E * DV);
        gemm(tij, wre[l], F(G.reb), gE, E, 128, 128, 1, 1, 1, 0);   // silu fused
        float* aE = allocf((long long)E * 8);
        k_sea_logits<<<gr((long long)E * 8), 256, 0, stream>>>(qn, kn, gE, nj, ni, aE, E);
        float* amax = allocf((long long)N * 8);
        float* asum = allocf((long long)N * 8);
        k_fill<<<gr((long long)N * 8), 256, 0, stream>>>(amax, -1e30f, (long long)N * 8);
        k_fill<<<gr((long long)N * 8), 256, 0, stream>>>(asum, 0.f, (long long)N * 8);
        k_seg_max<<<gr((long long)E * 8), 256, 0, stream>>>(aE, ni, amax, E);
        k_seg_exp_sum<<<gr((long long)E * 8), 256, 0, stream>>>(aE, amax, ni, asum, E);
        float* av = allocf((long long)E * SD);
        k_seg_norm_av<<<gr((long long)E * SD), 256, 0, stream>>>(aE, asum, ni, nj, vn, av, E, SD);
        float* sea = allocf((long long)E * SD);
        gemm(av, wcmb[l], F(G.cb), sea, E, SD, SD, 0, 1, 1, 0);

        // o = sea + w_rs(t) * mlp_s(h)[n_j] * cc   (assembled on the fly in scatters)
        float* rsb = allocf((long long)E * SD);
        gemm(tij, wrs[l], F(G.rsb), rsb, E, 128, SD, 0, 1, 1, 0);
        float* shh  = allocf((long long)N * DV);
        float* sfac = allocf((long long)N * SD);
        gemm(hln, ws1[l], F(G.s1b), shh, N, 128, 128, 1, 1, 1, 0);
        gemm(shh, ws2[l], F(G.s2b), sfac, N, 128, SD, 0, 1, 1, 0);

        float* hnew = allocf((long long)N * DV);
        k_copy<<<gr((long long)N * DV), 256, 0, stream>>>(hnew, hln, (long long)N * DV);
        k_scatter_h<<<gr((long long)E * DV), 256, 0, stream>>>(sea, rsb, sfac, ccv, nj, ni, hnew, E, SD);

        float* Xnew = allocf((long long)N * XCOMP * DV);
        if (l == 0)
            k_fill<<<gr((long long)N * XCOMP * DV), 256, 0, stream>>>(Xnew, 0.f, (long long)N * XCOMP * DV);
        else
            k_copy<<<gr((long long)N * XCOMP * DV), 256, 0, stream>>>(Xnew, X, (long long)N * XCOMP * DV);
        k_scatter_X<<<gr((long long)E * XCOMP * DV), 256, 0, stream>>>(
            sea, rsb, sfac, ccv, l1e, l2e, X, nj, ni, Xnew, E, SD, (l == 0) ? 1 : 0);
        h = hnew;
        X = Xnew;

        // eqff
        float* Xvu = allocf((long long)N * XCOMP * DV);
        gemm(X, wvu[l], nullptr, Xvu, N * 8, 128, 128, 0, 1, 1, 0);
        float* l2n = allocf((long long)N * DV);
        k_l2n<<<gr((long long)N * DV), 256, 0, stream>>>(Xvu, l2n, N);
        float* cat = allocf((long long)N * 256);
        k_cat<<<gr((long long)N * DV), 256, 0, stream>>>(h, l2n, cat, N);
        lnorm(cat, Q.lng, Q.lnbe, cat, N, 256, 0);
        float* mh = allocf((long long)N * DV);
        float* mm = allocf((long long)N * 256);
        gemm(cat, wm1[l], F(Q.m1b), mh, N, 256, 128, 1, 1, 1, 0);
        gemm(mh, wm2[l], F(Q.m2b), mm, N, 128, 256, 0, 1, 1, 0);
        k_add_h<<<gr((long long)N * DV), 256, 0, stream>>>(h, mm, (long long)N * DV);
        k_xupd<<<gr((long long)N * XCOMP * DV), 256, 0, stream>>>(X, mm, Xvu, (long long)N * XCOMP * DV);
    }

    // ---- decode ----
    float* dh = allocf((long long)N * DV);
    gemm(h, w_d1, F(6), dh, N, 128, 128, 1, 1, 1, 0);
    k_fill<<<1, 256, 0, stream>>>(out, 0.f, (long long)out_size);
    k_decode<<<gr(N), 256, 0, stream>>>(dh, F(9), F(8), batch, out, N);
}
